// SemiFixedBilateralFilter_38998303048131
// MI455X (gfx1250) — matrix operand order repaired and checked by reading, not compiled, not run on hardware
//
#include <hip/hip_runtime.h>

typedef __attribute__((ext_vector_type(16))) _Float16 v16h;
typedef __attribute__((ext_vector_type(8)))  _Float16 v8h;
typedef __attribute__((ext_vector_type(4)))  _Float16 v4h;
typedef __attribute__((ext_vector_type(8)))  float    v8f;
typedef __attribute__((ext_vector_type(2)))  float    v2f;

#define TX  16       // tile width (= WMMA M/N granularity)
#define TY  32       // tile height (amortizes halo: 2.4x vs 3.06x read amp)
#define HC  28       // halo cols = TX + 12
#define HR  44       // halo rows = TY + 12
#define GP  36       // gram pitch (floats): 8*36 % 64 == 32 -> halves split banks
#define LOG2E 1.4426950408889634f

__global__ __launch_bounds__(256) void bilateral_wmma_kernel(
    const float* __restrict__ in, const float* __restrict__ prm,
    float* __restrict__ out, int H, int W)
{
  __shared__ __align__(16) _Float16 xh[HR*HC*16];   // [pixel][channel] f16
  __shared__ __align__(16) _Float16 zpad[16];       // zero source for B pad
  __shared__ float sh_s[HR*HC];                     // s(q) (log2-scaled)
  __shared__ float psh[18];                         // params * log2(e)
  __shared__ float gram[8][16*GP];                  // per-wave, win1/win2 paired

  const int tid = threadIdx.x;
  const int b   = blockIdx.z;
  const int x0  = blockIdx.x * TX;
  const int y0  = blockIdx.y * TY;

  if (tid < 18) psh[tid] = prm[tid] * LOG2E;   // work in log2 domain
  if (tid < 16) zpad[tid] = (_Float16)0.f;
  __syncthreads();

  float pr[16];
  #pragma unroll
  for (int c = 0; c < 16; ++c) pr[c] = psh[c];

  // ---- preload halo (global -> LDS f16, plus per-pixel s) ----
  for (int pi = tid; pi < HR*HC; pi += 256) {
    int hr = pi / HC, hc = pi % HC;
    int gy = y0 - 6 + hr, gx = x0 - 6 + hc;
    bool v = (gy >= 0) && (gy < H) && (gx >= 0) && (gx < W);
    float s = 0.f;
    v8h e0, e1;
    #pragma unroll
    for (int c = 0; c < 8; ++c) {
      float x = v ? in[(((b*16 + c)*H) + gy)*W + gx] : 0.f;
      e0[c] = (_Float16)x;  s += pr[c] * x * x;
    }
    #pragma unroll
    for (int c = 0; c < 8; ++c) {
      float x = v ? in[(((b*16 + 8 + c)*H) + gy)*W + gx] : 0.f;
      e1[c] = (_Float16)x;  s += pr[8 + c] * x * x;
    }
    *(v8h*)&xh[pi*16]     = e0;
    *(v8h*)&xh[pi*16 + 8] = e1;
    sh_s[pi] = s;
  }
  __syncthreads();

  const int lane = tid & 31;
  const int wid  = tid >> 5;
  const int p    = lane & 15;   // pixel column within tile (and N index)
  const int h    = lane >> 4;   // wave half

  const float sx = psh[16], sy = psh[17];

  _Float16 ph[8];               // per-lane f16 (log2-scaled) range coeffs
  #pragma unroll
  for (int e = 0; e < 8; ++e) ph[e] = (_Float16)psh[8*h + e];

  // ---- per-(lane,tap) dy-invariants, hoisted out of all loops ----
  // h=0 -> dx in {-6,-4,-2,0}; h=1 -> dx in {2,4,6,(pad)}
  float sxdx2[4], mx[4];
  int   co[4], sqo[4], xqo[4];
  #pragma unroll
  for (int t = 0; t < 4; ++t) {
    const int dxi = 4*h + t;
    const int dx  = 2*(dxi - 3);
    const int jj  = p + dx + 6;                       // gram col (pre-split)
    const int jq  = (jj < HC) ? jj : (HC - 1);        // clamped halo col
    sxdx2[t] = sx * (float)(dx*dx);
    const int gxq = x0 + p + dx;
    mx[t] = ((dxi < 7) && (gxq >= 0) && (gxq < W)) ? 1.f : 0.f;
    co[t]  = p*GP + ((jj < 16) ? 2*jj : 2*jj - 31);   // win1 even / win2 odd
    sqo[t] = jq;
    xqo[t] = jq*16;
  }

  for (int rr = 0; rr < TY/8; ++rr) {
    const int ry = wid + rr*8;  // row within tile (uniform per wave)

    // A: 16 pixels x K(=16ch, zero-padded to 32) of a_c = k'_c * x_c(p)
    const int pbase = ((ry+6)*HC + (p+6))*16 + 8*h;
    const v8h xa = *(const v8h*)&xh[pbase];
    v16h A;
    #pragma unroll
    for (int e = 0; e < 8; ++e) { A[e] = ph[e] * xa[e]; A[8+e] = (_Float16)0.f; }

    const float sp = sh_s[(ry+6)*HC + (p+6)];
    float num0 = 0.f, num1 = 0.f, num2 = 0.f, den = 0.f;

    for (int dyi = 0; dyi < 7; ++dyi) {
      const int dy   = 2*(dyi - 3);
      const int qhr  = ry + 6 + dy;          // neighbor row in halo coords
      const int qrow = qhr*HC;

      // B: lane half h = channels 8h..8h+7 of window column p; elems 8-15 -> zero pad
      const bool l2 = (16 + p < HC);
      const _Float16* b1p = &xh[(qrow + p)*16 + 8*h];
      const _Float16* b2p = l2 ? &xh[(qrow + 16 + p)*16 + 8*h] : zpad;
      const v8h t0 = *(const v8h*)(b1p);
      const v8h t1 = *(const v8h*)(zpad);
      const v8h u0 = *(const v8h*)(b2p);
      const v8h u1 = *(const v8h*)(zpad);
      v16h B1, B2;
      #pragma unroll
      for (int e = 0; e < 8; ++e) {
        B1[e] = t0[e];  B1[8+e] = t1[e];
        B2[e] = u0[e];  B2[8+e] = u1[e];
      }

      const v8f c0 = {};
      v8f d1 = __builtin_amdgcn_wmma_f32_16x16x32_f16(false, A, false, B1,
                                                      (short)0, c0, false, false);
      v8f d2 = __builtin_amdgcn_wmma_f32_16x16x32_f16(false, A, false, B2,
                                                      (short)0, c0, false, false);

      // spill gram as paired {win1,win2} float2 -> 8x ds_store_b64
      #pragma unroll
      for (int r = 0; r < 8; ++r) {
        v2f st; st.x = d1[r]; st.y = d2[r];
        *(v2f*)&gram[wid][(r + 8*h)*GP + 2*p] = st;
      }

      const float dyterm = sy * (float)(dy*dy);
      const int   gyq    = y0 + ry + dy;
      const float my     = ((gyq >= 0) && (gyq < H)) ? 1.f : 0.f;

      #pragma unroll
      for (int t = 0; t < 4; ++t) {
        const float cross = gram[wid][co[t]];
        const float sq    = sh_s[qrow + sqo[t]];
        const float logw  = sp + sq - 2.f*cross + sxdx2[t] + dyterm;
        const float w     = (mx[t]*my) * exp2f(fminf(logw, 0.f));
        const v4h   xq    = *(const v4h*)&xh[qrow*16 + xqo[t]];
        den  += w;
        num0 += w * (float)xq[0];
        num1 += w * (float)xq[1];
        num2 += w * (float)xq[2];
      }
    }

    den  += __shfl_xor(den,  16, 32);
    num0 += __shfl_xor(num0, 16, 32);
    num1 += __shfl_xor(num1, 16, 32);
    num2 += __shfl_xor(num2, 16, 32);

    if (lane < 16) {
      const int gy = y0 + ry, gx = x0 + p;
      const float invd = 1.f / den;
      out[(((b*3 + 0)*H) + gy)*W + gx] = num0 * invd;
      out[(((b*3 + 1)*H) + gy)*W + gx] = num1 * invd;
      out[(((b*3 + 2)*H) + gy)*W + gx] = num2 * invd;
    }
  }
}

extern "C" void kernel_launch(void* const* d_in, const int* in_sizes, int n_in,
                              void* d_out, int out_size, void* d_ws, size_t ws_size,
                              hipStream_t stream) {
  (void)in_sizes; (void)n_in; (void)d_ws; (void)ws_size; (void)out_size;
  const float* in  = (const float*)d_in[0];
  const float* prm = (const float*)d_in[1];
  float* out = (float*)d_out;
  const int H = 512, W = 512, B = 4;
  dim3 grid(W / TX, H / TY, B);
  bilateral_wmma_kernel<<<grid, 256, 0, stream>>>(in, prm, out, H, W);
}
